// PatchDiffusionDiT_90486370992152
// MI455X (gfx1250) — matrix operand order repaired and checked by reading, not compiled, not run on hardware
//
#include <hip/hip_runtime.h>
#include <hip/hip_bf16.h>

// ---------------------------------------------------------------------------
// Types for CDNA5 WMMA (wave32, 16x16x32 bf16 -> f32)
// ---------------------------------------------------------------------------
typedef __bf16 bf16_t;
typedef __attribute__((ext_vector_type(16))) __bf16 v16bf;
typedef __attribute__((ext_vector_type(8)))  __bf16 v8bf;
typedef __attribute__((ext_vector_type(8)))  float  v8f;

#define DEV __device__ __forceinline__

DEV float bf2f(bf16_t h) {
    unsigned short u = __builtin_bit_cast(unsigned short, h);
    unsigned v = ((unsigned)u) << 16;
    return __builtin_bit_cast(float, v);
}
DEV bf16_t f2bf(float f) {
    unsigned u = __builtin_bit_cast(unsigned, f);
    u = (u + 0x7FFFu + ((u >> 16) & 1u)) >> 16;
    return __builtin_bit_cast(bf16_t, (unsigned short)u);
}
DEV v8f vzero8() { v8f z = {0.f,0.f,0.f,0.f,0.f,0.f,0.f,0.f}; return z; }

DEV float siluf(float x) { return x / (1.0f + __expf(-x)); }

// A-fragment (16x32 bf16) from row-major memory. Lane l<16: row l, K {0..7,16..23};
// lane>=16: row l-16, K {8..15,24..31}. Two contiguous 16B loads per lane.
DEV v16bf load_a_frag(const bf16_t* A, int lda) {
    int lane = (int)(threadIdx.x & 31u);
    const bf16_t* p = A + (long)(lane & 15) * lda + ((lane < 16) ? 0 : 8);
    union { v16bf v; v8bf h[2]; } u;
    u.h[0] = *(const v8bf*)p;
    u.h[1] = *(const v8bf*)(p + 16);
    return u.v;
}
// B-fragment (32x16 bf16) from a transposed matrix Bt (N x K, row-major):
// lane l<16: column l, K {0..7,16..23}; lane>=16: column l-16, K {8..15,24..31}.
DEV v16bf load_b_frag(const bf16_t* Bt, int ldb) {
    int lane = (int)(threadIdx.x & 31u);
    const bf16_t* p = Bt + (long)(lane & 15) * ldb + ((lane < 16) ? 0 : 8);
    union { v16bf v; v8bf h[2]; } u;
    u.h[0] = *(const v8bf*)p;
    u.h[1] = *(const v8bf*)(p + 16);
    return u.v;
}
DEV v8f wmma_bf16(v16bf a, v16bf b, v8f c) {
    return __builtin_amdgcn_wmma_f32_16x16x32_bf16(false, a, false, b, (short)0, c, false, false);
}

// Model dims
#define BB   8
#define NN   256
#define DD   768
#define NHH  12
#define HDD  64
#define EE   8
#define HEE  2048
#define TOK  2048           // B*N
#define SLOTS 4096          // TOK * top_k

// ---------------------------------------------------------------------------
// Weight pack: W (nmat of K x N, f32 row-major) -> Wt (nmat of N x K, bf16)
// ---------------------------------------------------------------------------
__global__ void k_packT(const float* __restrict__ W, bf16_t* __restrict__ Wt,
                        int K, int N, int nmat) {
    long idx = (long)blockIdx.x * 256 + threadIdx.x;
    long per = (long)K * N;
    long tot = per * nmat;
    if (idx >= tot) return;
    int m = (int)(idx / per);
    long r = idx % per;
    int n = (int)(r / K);
    int k = (int)(r % K);
    Wt[idx] = f2bf(W[(long)m * per + (long)k * N + n]);
}

// ---------------------------------------------------------------------------
// Patchify: x (B,3,256,256) -> xp bf16 (B*N, 768), feature = c*256 + pi*16 + pj
// ---------------------------------------------------------------------------
__global__ void k_patchify(const float* __restrict__ x, bf16_t* __restrict__ xp) {
    long idx = (long)blockIdx.x * 256 + threadIdx.x;
    if (idx >= (long)TOK * DD) return;
    int f = (int)(idx % DD);
    long bn = idx / DD;
    int n = (int)(bn % NN);
    int b = (int)(bn / NN);
    int c = f / 256;
    int pi = (f % 256) / 16;
    int pj = f % 16;
    int gy = n / 16, gx = n % 16;
    long src = (((long)b * 3 + c) * 256 + gy * 16 + pi) * 256 + (gx * 16 + pj);
    xp[idx] = f2bf(x[src]);
}

// ---------------------------------------------------------------------------
// Generic WMMA GEMM: C(MxN) = A(MxK bf16 row-major) * B (Bt = NxK bf16)
// Block 256 thr = 8 waves (4M x 2N), block tile 128x128, wave tile 32x64.
// epi: 0 = f32 (+bias), 1 = bias+GELU -> bf16, 2 = resid += gate*acc,
//      3 = gathered scatter: Cf[rowidx[m]] = acc*slotw[rowidx[m]]
// ---------------------------------------------------------------------------
__global__ __launch_bounds__(256) void k_gemm(
    const bf16_t* __restrict__ A, int lda,
    const bf16_t* __restrict__ Bt, long btStrideZ,
    int M, int N, int K,
    const int* __restrict__ Mcnt,
    const int* __restrict__ rowidx, int ridxStrideZ,
    float* __restrict__ Cf, bf16_t* __restrict__ Cb, int ldc,
    const float* __restrict__ bias, int epi,
    const float* __restrict__ mod, int modStride, int gateOff,
    float* __restrict__ resid, const float* __restrict__ slotw) {
    int z = blockIdx.z;
    int Me = Mcnt ? Mcnt[z] : M;
    int m0 = blockIdx.y * 128;
    if (m0 >= Me) return;
    int n0 = blockIdx.x * 128;
    const bf16_t* B = Bt + (long)z * btStrideZ;
    const int* ridx = rowidx ? (rowidx + (long)z * ridxStrideZ) : nullptr;

    int wid = (int)(threadIdx.x >> 5);
    int lane = (int)(threadIdx.x & 31u);
    int wm = wid & 3, wn = wid >> 2;
    int wrow = m0 + wm * 32;
    int wcol = n0 + wn * 64;

    int ar0 = wrow + (lane & 15);
    int ar1 = ar0 + 16;
    long aoff0, aoff1;
    if (ridx) {
        int i0 = (ar0 < Me) ? ridx[ar0] : 0;
        int i1 = (ar1 < Me) ? ridx[ar1] : 0;
        aoff0 = (long)i0 * lda;
        aoff1 = (long)i1 * lda;
    } else {
        aoff0 = (long)ar0 * lda;
        aoff1 = (long)ar1 * lda;
    }
    int kbase = (lane < 16) ? 0 : 8;

    v8f acc[2][4];
#pragma unroll
    for (int mi = 0; mi < 2; mi++)
#pragma unroll
        for (int nf = 0; nf < 4; nf++) acc[mi][nf] = vzero8();

    for (int k0 = 0; k0 < K; k0 += 32) {
        // speculative prefetch of next K tile (gfx1250 global_prefetch path)
        if (k0 + 32 < K) {
            __builtin_prefetch(A + aoff0 + k0 + 32, 0, 1);
            __builtin_prefetch(B + (long)(wcol + (lane & 3) * 16) * K + k0 + 32, 0, 1);
        }
        union { v16bf v; v8bf h[2]; } a0, a1;
        const bf16_t* pa0 = A + aoff0 + k0 + kbase;
        const bf16_t* pa1 = A + aoff1 + k0 + kbase;
        a0.h[0] = *(const v8bf*)pa0;  a0.h[1] = *(const v8bf*)(pa0 + 16);
        a1.h[0] = *(const v8bf*)pa1;  a1.h[1] = *(const v8bf*)(pa1 + 16);
        v16bf bfr[4];
#pragma unroll
        for (int nf = 0; nf < 4; nf++)
            bfr[nf] = load_b_frag(B + (long)(wcol + nf * 16) * K + k0, K);
#pragma unroll
        for (int nf = 0; nf < 4; nf++) {
            acc[0][nf] = wmma_bf16(a0.v, bfr[nf], acc[0][nf]);
            acc[1][nf] = wmma_bf16(a1.v, bfr[nf], acc[1][nf]);
        }
    }

    int cn = lane & 15;
    int rb = (lane < 16) ? 0 : 8;
#pragma unroll
    for (int mi = 0; mi < 2; mi++) {
#pragma unroll
        for (int r = 0; r < 8; r++) {
            int mrow = wrow + mi * 16 + rb + r;
            if (mrow >= Me) continue;
            long orow = ridx ? (long)ridx[mrow] : (long)mrow;
#pragma unroll
            for (int nf = 0; nf < 4; nf++) {
                int col = wcol + nf * 16 + cn;
                float v = acc[mi][nf][r];
                if (bias) v += bias[col];
                if (epi == 0) {
                    Cf[orow * ldc + col] = v;
                } else if (epi == 1) {
                    float g = 0.5f * v * (1.0f + erff(v * 0.70710678f));
                    Cb[orow * ldc + col] = f2bf(g);
                } else if (epi == 2) {
                    int bb = mrow >> 8;
                    float gate = mod[(long)bb * modStride + gateOff + col];
                    resid[orow * ldc + col] += gate * v;
                } else {
                    Cf[orow * ldc + col] = v * slotw[orow];
                }
            }
        }
    }
}

// ---------------------------------------------------------------------------
// MoE GEMM1 (gathered, dual-accumulator): act[slot] = silu(x W1) * (x W3)
// grid: (HE/128, 32, E). A rows gathered via rowlist (token = slot>>1).
// ---------------------------------------------------------------------------
__global__ __launch_bounds__(256) void k_moe_gemm1(
    const bf16_t* __restrict__ A,
    const bf16_t* __restrict__ W1t, const bf16_t* __restrict__ W3t,
    const int* __restrict__ count, const int* __restrict__ rowlist,
    bf16_t* __restrict__ act) {
    int e = blockIdx.z;
    int Me = count[e];
    int m0 = blockIdx.y * 128;
    if (m0 >= Me) return;
    int n0 = blockIdx.x * 128;
    const bf16_t* B1 = W1t + (long)e * HEE * DD;
    const bf16_t* B3 = W3t + (long)e * HEE * DD;
    const int* ridx = rowlist + (long)e * SLOTS;

    int wid = (int)(threadIdx.x >> 5);
    int lane = (int)(threadIdx.x & 31u);
    int wm = wid & 3, wn = wid >> 2;
    int wrow = m0 + wm * 32;
    int wcol = n0 + wn * 64;

    int ar0 = wrow + (lane & 15);
    int ar1 = ar0 + 16;
    long aoff0 = (long)(((ar0 < Me) ? ridx[ar0] : 0) >> 1) * DD;
    long aoff1 = (long)(((ar1 < Me) ? ridx[ar1] : 0) >> 1) * DD;
    int kbase = (lane < 16) ? 0 : 8;

    v8f a1c[2][4], a3c[2][4];
#pragma unroll
    for (int mi = 0; mi < 2; mi++)
#pragma unroll
        for (int nf = 0; nf < 4; nf++) { a1c[mi][nf] = vzero8(); a3c[mi][nf] = vzero8(); }

    for (int k0 = 0; k0 < DD; k0 += 32) {
        if (k0 + 32 < DD) {
            __builtin_prefetch(A + aoff0 + k0 + 32, 0, 1);
            __builtin_prefetch(B1 + (long)(wcol + (lane & 3) * 16) * DD + k0 + 32, 0, 1);
            __builtin_prefetch(B3 + (long)(wcol + (lane & 3) * 16) * DD + k0 + 32, 0, 1);
        }
        union { v16bf v; v8bf h[2]; } a0, a1;
        const bf16_t* pa0 = A + aoff0 + k0 + kbase;
        const bf16_t* pa1 = A + aoff1 + k0 + kbase;
        a0.h[0] = *(const v8bf*)pa0;  a0.h[1] = *(const v8bf*)(pa0 + 16);
        a1.h[0] = *(const v8bf*)pa1;  a1.h[1] = *(const v8bf*)(pa1 + 16);
#pragma unroll
        for (int nf = 0; nf < 4; nf++) {
            v16bf b1 = load_b_frag(B1 + (long)(wcol + nf * 16) * DD + k0, DD);
            v16bf b3 = load_b_frag(B3 + (long)(wcol + nf * 16) * DD + k0, DD);
            a1c[0][nf] = wmma_bf16(a0.v, b1, a1c[0][nf]);
            a1c[1][nf] = wmma_bf16(a1.v, b1, a1c[1][nf]);
            a3c[0][nf] = wmma_bf16(a0.v, b3, a3c[0][nf]);
            a3c[1][nf] = wmma_bf16(a1.v, b3, a3c[1][nf]);
        }
    }

    int cn = lane & 15;
    int rb = (lane < 16) ? 0 : 8;
#pragma unroll
    for (int mi = 0; mi < 2; mi++) {
#pragma unroll
        for (int r = 0; r < 8; r++) {
            int mrow = wrow + mi * 16 + rb + r;
            if (mrow >= Me) continue;
            long slot = (long)ridx[mrow];
#pragma unroll
            for (int nf = 0; nf < 4; nf++) {
                int col = wcol + nf * 16 + cn;
                float v1 = a1c[mi][nf][r];
                float v3 = a3c[mi][nf][r];
                act[slot * HEE + col] = f2bf(siluf(v1) * v3);
            }
        }
    }
}

// ---------------------------------------------------------------------------
// Flash attention: one block (8 waves) per (b,h); wave = 32 query rows.
// Online softmax over 64-key chunks; P bounced via LDS for layout change.
// q,k: (B*NH, 256, 64) bf16 ; vt: (B*NH, 64, 256) bf16 ; o: (B*N, 768) bf16
// ---------------------------------------------------------------------------
__global__ __launch_bounds__(256) void k_attn(
    const bf16_t* __restrict__ q, const bf16_t* __restrict__ kk,
    const bf16_t* __restrict__ vt, bf16_t* __restrict__ o) {
    __shared__ bf16_t Plds[256 * 64];
    int bh = blockIdx.x;
    int b  = bh / NHH;
    int hd = bh % NHH;
    const bf16_t* qb = q  + (long)bh * NN * HDD;
    const bf16_t* kb = kk + (long)bh * NN * HDD;
    const bf16_t* vb = vt + (long)bh * HDD * NN;
    int wid = (int)(threadIdx.x >> 5);
    int lane = (int)(threadIdx.x & 31u);
    int rowbase = wid * 32;

    v16bf qf[2][2];
#pragma unroll
    for (int mi = 0; mi < 2; mi++)
#pragma unroll
        for (int ks = 0; ks < 2; ks++)
            qf[mi][ks] = load_a_frag(qb + (long)(rowbase + mi * 16) * HDD + ks * 32, HDD);

    v8f oacc[2][4];
    float mrow[2][8], lrow[2][8];
#pragma unroll
    for (int mi = 0; mi < 2; mi++) {
#pragma unroll
        for (int df = 0; df < 4; df++) oacc[mi][df] = vzero8();
#pragma unroll
        for (int r = 0; r < 8; r++) { mrow[mi][r] = -1e30f; lrow[mi][r] = 0.f; }
    }

    for (int j0 = 0; j0 < NN; j0 += 64) {
        if (j0 + 64 < NN) {
            __builtin_prefetch(kb + (long)(j0 + 64 + (lane & 3) * 16) * HDD, 0, 1);
            __builtin_prefetch(vb + (long)((lane & 3) * 16) * NN + j0 + 64, 0, 1);
        }
        v8f s[2][4];
#pragma unroll
        for (int mi = 0; mi < 2; mi++)
#pragma unroll
            for (int nf = 0; nf < 4; nf++) s[mi][nf] = vzero8();
#pragma unroll
        for (int ks = 0; ks < 2; ks++) {
            v16bf bfr[4];
#pragma unroll
            for (int nf = 0; nf < 4; nf++)
                bfr[nf] = load_b_frag(kb + (long)(j0 + nf * 16) * HDD + ks * 32, HDD);
#pragma unroll
            for (int mi = 0; mi < 2; mi++)
#pragma unroll
                for (int nf = 0; nf < 4; nf++)
                    s[mi][nf] = wmma_bf16(qf[mi][ks], bfr[nf], s[mi][nf]);
        }
        // scale + online softmax update (per row: reg index r within 16-lane group)
#pragma unroll
        for (int mi = 0; mi < 2; mi++) {
#pragma unroll
            for (int r = 0; r < 8; r++) {
                float mx = -1e30f;
#pragma unroll
                for (int nf = 0; nf < 4; nf++) {
                    s[mi][nf][r] *= 0.125f;
                    mx = fmaxf(mx, s[mi][nf][r]);
                }
#pragma unroll
                for (int d = 1; d < 16; d <<= 1) mx = fmaxf(mx, __shfl_xor(mx, d, 32));
                float nm = fmaxf(mrow[mi][r], mx);
                float al = __expf(mrow[mi][r] - nm);
                mrow[mi][r] = nm;
                float ls = 0.f;
#pragma unroll
                for (int nf = 0; nf < 4; nf++) {
                    float p = __expf(s[mi][nf][r] - nm);
                    s[mi][nf][r] = p;
                    ls += p;
                }
#pragma unroll
                for (int d = 1; d < 16; d <<= 1) ls += __shfl_xor(ls, d, 32);
                lrow[mi][r] = lrow[mi][r] * al + ls;
#pragma unroll
                for (int df = 0; df < 4; df++) oacc[mi][df][r] *= al;
            }
        }
        // spill P (bf16) to this wave's private LDS slice, then reload as A-frags
        {
            int cn = lane & 15;
            int rb = (lane < 16) ? 0 : 8;
#pragma unroll
            for (int mi = 0; mi < 2; mi++)
#pragma unroll
                for (int nf = 0; nf < 4; nf++)
#pragma unroll
                    for (int r = 0; r < 8; r++) {
                        int row = rowbase + mi * 16 + rb + r;
                        Plds[row * 64 + nf * 16 + cn] = f2bf(s[mi][nf][r]);
                    }
        }
        __syncthreads();
#pragma unroll
        for (int ks = 0; ks < 2; ks++) {
            v16bf pa[2];
#pragma unroll
            for (int mi = 0; mi < 2; mi++)
                pa[mi] = load_a_frag(&Plds[(rowbase + mi * 16) * 64 + ks * 32], 64);
#pragma unroll
            for (int df = 0; df < 4; df++) {
                v16bf vf = load_b_frag(vb + (long)(df * 16) * NN + j0 + ks * 32, NN);
#pragma unroll
                for (int mi = 0; mi < 2; mi++)
                    oacc[mi][df] = wmma_bf16(pa[mi], vf, oacc[mi][df]);
            }
        }
        __syncthreads();
    }
    // normalize and write o (row-major B*N x 768)
    {
        int cn = lane & 15;
        int rb = (lane < 16) ? 0 : 8;
#pragma unroll
        for (int mi = 0; mi < 2; mi++)
#pragma unroll
            for (int r = 0; r < 8; r++) {
                int row = rowbase + mi * 16 + rb + r;
                long orow = (long)b * NN + row;
                float inv = 1.0f / lrow[mi][r];
#pragma unroll
                for (int df = 0; df < 4; df++)
                    o[orow * DD + hd * HDD + df * 16 + cn] = f2bf(oacc[mi][df][r] * inv);
            }
    }
}

// ---------------------------------------------------------------------------
// RMSNorm + adaLN modulate -> bf16:  out = rms(x,w)*(1+mod[scale]) + mod[shift]
// one block (256 thr) per token row
// ---------------------------------------------------------------------------
__global__ void k_rmsmod(const float* __restrict__ x, const float* __restrict__ w,
                         const float* __restrict__ mod, int modStride,
                         int shiftOff, int scaleOff, bf16_t* __restrict__ out) {
    __shared__ float red[8];
    int row = blockIdx.x;
    int b = row >> 8;
    int tid = (int)threadIdx.x;
    const float* xr = x + (long)row * DD;
    float ss = 0.f;
    for (int i = tid; i < DD; i += 256) { float v = xr[i]; ss += v * v; }
#pragma unroll
    for (int d = 1; d < 32; d <<= 1) ss += __shfl_xor(ss, d, 32);
    if ((tid & 31) == 0) red[tid >> 5] = ss;
    __syncthreads();
    float tot = red[0] + red[1] + red[2] + red[3] + red[4] + red[5] + red[6] + red[7];
    float rs = rsqrtf(tot * (1.0f / DD) + 1e-6f);
    const float* mr = mod + (long)b * modStride;
    for (int i = tid; i < DD; i += 256) {
        float v = xr[i] * rs * w[i];
        out[(long)row * DD + i] = f2bf(v * (1.0f + mr[scaleOff + i]) + mr[shiftOff + i]);
    }
}

// ---------------------------------------------------------------------------
// QKV prep: per (b,n,h): head RMSNorm on q,k + 2D RoPE; write q,k row-major
// bf16 and v transposed bf16. blockDim = 64 (one head-row).
// ---------------------------------------------------------------------------
__global__ void k_qkvprep(const float* __restrict__ qkv,
                          const float* __restrict__ qnw, const float* __restrict__ knw,
                          bf16_t* __restrict__ qo, bf16_t* __restrict__ ko,
                          bf16_t* __restrict__ vo) {
    __shared__ float red[64];
    __shared__ float qs[64];
    __shared__ float ks_[64];
    int id = blockIdx.x;               // (b*256+n)*12 + h
    int h  = id % NHH;
    int bn = id / NHH;
    int n  = bn % NN;
    int b  = bn / NN;
    int d  = (int)threadIdx.x;
    long base = (long)bn * (3 * DD) + h * HDD + d;
    float qv = qkv[base];
    float kv = qkv[base + DD];
    float vv = qkv[base + 2 * DD];

    red[d] = qv * qv; __syncthreads();
    for (int s1 = 32; s1 > 0; s1 >>= 1) { if (d < s1) red[d] += red[d + s1]; __syncthreads(); }
    float qrs = rsqrtf(red[0] * (1.0f / HDD) + 1e-6f);
    __syncthreads();
    red[d] = kv * kv; __syncthreads();
    for (int s1 = 32; s1 > 0; s1 >>= 1) { if (d < s1) red[d] += red[d + s1]; __syncthreads(); }
    float krs = rsqrtf(red[0] * (1.0f / HDD) + 1e-6f);
    __syncthreads();

    float qn = qv * qrs * qnw[d];
    float kn = kv * krs * knw[d];
    qs[d] = qn; ks_[d] = kn;
    __syncthreads();

    float pos = (d < 32) ? (float)(n / 16) : (float)(n % 16);
    int j = (d < 32) ? (d >> 1) : ((d - 32) >> 1);
    float ang = pos * __powf(10000.0f, -(float)j * (1.0f / 16.0f));
    float cc = __cosf(ang), sn = __sinf(ang);
    float qr = (d & 1) ? qs[d - 1] : -qs[d + 1];
    float kr = (d & 1) ? ks_[d - 1] : -ks_[d + 1];
    float qe = qn * cc + qr * sn;
    float ke = kn * cc + kr * sn;

    long hb = (long)(b * NHH + h) * NN + n;
    qo[hb * HDD + d] = f2bf(qe);
    ko[hb * HDD + d] = f2bf(ke);
    vo[((long)(b * NHH + h) * HDD + d) * NN + n] = f2bf(vv);
}

// ---------------------------------------------------------------------------
// Router: one wave per token; top-2 + softmax gates; build per-expert lists.
// ---------------------------------------------------------------------------
__global__ void k_router(const bf16_t* __restrict__ hh, const float* __restrict__ rw,
                         int* __restrict__ count, int* __restrict__ rowlist,
                         float* __restrict__ slotw) {
    int t = blockIdx.x * 8 + (int)(threadIdx.x >> 5);
    int lane = (int)(threadIdx.x & 31u);
    if (t >= TOK) return;
    float acc[EE];
#pragma unroll
    for (int e = 0; e < EE; e++) acc[e] = 0.f;
    const bf16_t* xr = hh + (long)t * DD;
    for (int i = lane; i < DD; i += 32) {
        float v = bf2f(xr[i]);
#pragma unroll
        for (int e = 0; e < EE; e++) acc[e] += v * rw[(long)i * EE + e];
    }
#pragma unroll
    for (int d = 1; d < 32; d <<= 1)
#pragma unroll
        for (int e = 0; e < EE; e++) acc[e] += __shfl_xor(acc[e], d, 32);
    if (lane == 0) {
        int i0 = 0; float v0 = acc[0];
#pragma unroll
        for (int e = 1; e < EE; e++) if (acc[e] > v0) { v0 = acc[e]; i0 = e; }
        int i1 = (i0 == 0) ? 1 : 0; float v1 = acc[i1];
#pragma unroll
        for (int e = 0; e < EE; e++) if (e != i0 && acc[e] > v1) { v1 = acc[e]; i1 = e; }
        float w0 = 1.0f / (1.0f + __expf(v1 - v0));
        slotw[t * 2]     = w0;
        slotw[t * 2 + 1] = 1.0f - w0;
        int p0 = atomicAdd(&count[i0], 1);
        rowlist[i0 * SLOTS + p0] = t * 2;
        int p1 = atomicAdd(&count[i1], 1);
        rowlist[i1 * SLOTS + p1] = t * 2 + 1;
    }
}

// h[t,col] += gp * (partial[2t,col] + partial[2t+1,col])  (partials pre-weighted)
__global__ void k_moecombine(float* __restrict__ h, const float* __restrict__ partial,
                             const float* __restrict__ mod) {
    long idx = (long)blockIdx.x * 256 + threadIdx.x;
    if (idx >= (long)TOK * DD) return;
    int col = (int)(idx % DD);
    long t = idx / DD;
    int b = (int)(t >> 8);
    float gp = mod[(long)b * (6 * DD) + 5 * DD + col];
    h[idx] += gp * (partial[t * 2 * DD + col] + partial[(t * 2 + 1) * DD + col]);
}

// mod = silu(c) @ W + bias   (M=8, K=768, N variable) — tiny, plain VALU
__global__ void k_modmlp(const float* __restrict__ c, const float* __restrict__ w,
                         const float* __restrict__ bias, float* __restrict__ out, int N) {
    int j = blockIdx.x * 256 + (int)threadIdx.x;
    int b = blockIdx.y;
    if (j >= N) return;
    const float* cr = c + (long)b * DD;
    float acc = 0.f;
    for (int i = 0; i < DD; i++) acc += siluf(cr[i]) * w[(long)i * N + j];
    out[(long)b * N + j] = acc + bias[j];
}

// temb1 = silu(te @ t1_w + t1_b), te computed on the fly from t
__global__ void k_temb1(const float* __restrict__ t, const float* __restrict__ w,
                        const float* __restrict__ bias, float* __restrict__ out) {
    int j = blockIdx.x * 256 + (int)threadIdx.x;
    int b = blockIdx.y;
    if (j >= DD) return;
    float tv = t[b];
    float acc = 0.f;
    const float L10K = 9.210340371976184f; // ln(10000)
    for (int i = 0; i < 256; i++) {
        int fi = (i < 128) ? i : (i - 128);
        float fr = __expf(-L10K * (float)fi * (1.0f / 128.0f));
        float te = (i < 128) ? __cosf(tv * fr) : __sinf(tv * fr);
        acc += te * w[(long)i * DD + j];
    }
    acc += bias[j];
    out[(long)b * DD + j] = siluf(acc);
}

// c = temb1 @ t2_w + t2_b + task_tab[task] + mod_tab[mod]
__global__ void k_temb2(const float* __restrict__ a, const float* __restrict__ w,
                        const float* __restrict__ bias, const int* __restrict__ task_id,
                        const int* __restrict__ mod_id, const float* __restrict__ task_tab,
                        const float* __restrict__ mod_tab, float* __restrict__ c) {
    int j = blockIdx.x * 256 + (int)threadIdx.x;
    int b = blockIdx.y;
    if (j >= DD) return;
    float acc = 0.f;
    const float* ar = a + (long)b * DD;
    for (int i = 0; i < DD; i++) acc += ar[i] * w[(long)i * DD + j];
    c[(long)b * DD + j] = acc + bias[j] + task_tab[(long)task_id[b] * DD + j]
                               + mod_tab[(long)mod_id[b] * DD + j];
}

__global__ void k_zero32(int* __restrict__ p, int n) {
    int i = blockIdx.x * 256 + (int)threadIdx.x;
    if (i < n) p[i] = 0;
}

// ---------------------------------------------------------------------------
// Host orchestration
// ---------------------------------------------------------------------------
static inline size_t alignup(size_t x) { return (x + 255) & ~(size_t)255; }

extern "C" void kernel_launch(void* const* d_in, const int* in_sizes, int n_in,
                              void* d_out, int out_size, void* d_ws, size_t ws_size,
                              hipStream_t stream) {
    (void)in_sizes; (void)n_in; (void)out_size; (void)ws_size;
    const float* x        = (const float*)d_in[0];
    const float* t_in     = (const float*)d_in[1];
    const int*   task_id  = (const int*)d_in[2];
    const int*   mod_id   = (const int*)d_in[3];
    const float* pe1_w    = (const float*)d_in[4];
    const float* pe1_b    = (const float*)d_in[5];
    const float* pe2_w    = (const float*)d_in[6];
    const float* pe2_b    = (const float*)d_in[7];
    const float* t1_w     = (const float*)d_in[8];
    const float* t1_b     = (const float*)d_in[9];
    const float* t2_w     = (const float*)d_in[10];
    const float* t2_b     = (const float*)d_in[11];
    const float* task_tab = (const float*)d_in[12];
    const float* mod_tab  = (const float*)d_in[13];
    const float* qkv_w    = (const float*)d_in[14];
    const float* proj_w   = (const float*)d_in[15];
    const float* qn_w     = (const float*)d_in[16];
    const float* kn_w     = (const float*)d_in[17];
    const float* norm1_w  = (const float*)d_in[18];
    const float* norm2_w  = (const float*)d_in[19];
    const float* adaln_w  = (const float*)d_in[20];
    const float* adaln_b  = (const float*)d_in[21];
    const float* router_w = (const float*)d_in[22];
    const float* moe_w1   = (const float*)d_in[23];
    const float* moe_w2   = (const float*)d_in[24];
    const float* moe_w3   = (const float*)d_in[25];
    const float* fnorm_w  = (const float*)d_in[26];
    const float* fada_w   = (const float*)d_in[27];
    const float* fada_b   = (const float*)d_in[28];
    const float* fproj_w  = (const float*)d_in[29];
    const float* fproj_b  = (const float*)d_in[30];
    float* out = (float*)d_out;

    char* w = (char*)d_ws;
    size_t off = 0;
    auto alloc = [&](size_t bytes) { void* p = w + off; off = alignup(off + bytes); return p; };

    bf16_t* xp_bf  = (bf16_t*)alloc((size_t)TOK * DD * 2);
    bf16_t* pemid  = (bf16_t*)alloc((size_t)TOK * 128 * 2);
    float*  hbuf   = (float*) alloc((size_t)TOK * DD * 4);
    float*  temb1  = (float*) alloc((size_t)BB * DD * 4);
    float*  cbuf   = (float*) alloc((size_t)BB * DD * 4);
    float*  modb   = (float*) alloc((size_t)BB * 6 * DD * 4);
    float*  fmodb  = (float*) alloc((size_t)BB * 2 * DD * 4);
    bf16_t* hh     = (bf16_t*)alloc((size_t)TOK * DD * 2);
    float*  qkvf   = (float*) alloc((size_t)TOK * 3 * DD * 4);
    bf16_t* qb     = (bf16_t*)alloc((size_t)BB * NHH * NN * HDD * 2);
    bf16_t* kb     = (bf16_t*)alloc((size_t)BB * NHH * NN * HDD * 2);
    bf16_t* vtb    = (bf16_t*)alloc((size_t)BB * NHH * HDD * NN * 2);
    bf16_t* ob     = (bf16_t*)alloc((size_t)TOK * DD * 2);
    bf16_t* qkvT   = (bf16_t*)alloc((size_t)3 * DD * DD * 2);
    bf16_t* projT  = (bf16_t*)alloc((size_t)DD * DD * 2);
    bf16_t* w1T    = (bf16_t*)alloc((size_t)EE * HEE * DD * 2);
    bf16_t* w3T    = (bf16_t*)alloc((size_t)EE * HEE * DD * 2);
    bf16_t* w2T    = (bf16_t*)alloc((size_t)EE * DD * HEE * 2);
    bf16_t* actb   = (bf16_t*)alloc((size_t)SLOTS * HEE * 2);
    float*  partial= (float*) alloc((size_t)SLOTS * DD * 4);
    int*    cnt    = (int*)   alloc((size_t)EE * 4);
    int*    rowlist= (int*)   alloc((size_t)EE * SLOTS * 4);
    float*  slotw  = (float*) alloc((size_t)SLOTS * 4);
    bf16_t* pe1T   = (bf16_t*)alloc((size_t)128 * DD * 2);
    bf16_t* pe2T   = (bf16_t*)alloc((size_t)DD * 128 * 2);
    bf16_t* fpT    = (bf16_t*)alloc((size_t)DD * DD * 2);

    auto gemm = [&](const bf16_t* A, int lda, const bf16_t* Bt, long bz,
                    int M, int N, int K, const int* Mcnt, const int* ridx, int rstride,
                    float* Cf, bf16_t* Cb, int ldc, const float* bias, int epi,
                    const float* mod, int modStride, int gateOff, float* resid,
                    const float* sw, int gz, int mtiles) {
        dim3 grid(N / 128, mtiles, gz);
        k_gemm<<<grid, 256, 0, stream>>>(A, lda, Bt, bz, M, N, K, Mcnt, ridx, rstride,
                                         Cf, Cb, ldc, bias, epi, mod, modStride, gateOff,
                                         resid, sw);
    };
    auto packT = [&](const float* W, bf16_t* Wt, int K, int N, int nmat) {
        long tot = (long)K * N * nmat;
        k_packT<<<dim3((unsigned)((tot + 255) / 256)), 256, 0, stream>>>(W, Wt, K, N, nmat);
    };

    // ---- patch embed ----
    k_patchify<<<dim3((TOK * DD) / 256), 256, 0, stream>>>(x, xp_bf);
    packT(pe1_w, pe1T, DD, 128, 1);
    packT(pe2_w, pe2T, 128, DD, 1);
    packT(fproj_w, fpT, DD, DD, 1);
    gemm(xp_bf, DD, pe1T, 0, TOK, 128, DD, nullptr, nullptr, 0,
         nullptr, pemid, 128, pe1_b, 1, nullptr, 0, 0, nullptr, nullptr, 1, TOK / 128);
    gemm(pemid, 128, pe2T, 0, TOK, DD, 128, nullptr, nullptr, 0,
         hbuf, nullptr, DD, pe2_b, 0, nullptr, 0, 0, nullptr, nullptr, 1, TOK / 128);

    // ---- conditioning ----
    k_temb1<<<dim3(3, BB), 256, 0, stream>>>(t_in, t1_w, t1_b, temb1);
    k_temb2<<<dim3(3, BB), 256, 0, stream>>>(temb1, t2_w, t2_b, task_id, mod_id,
                                             task_tab, mod_tab, cbuf);

    // ---- transformer blocks ----
    for (int l = 0; l < 4; l++) {
        k_modmlp<<<dim3(18, BB), 256, 0, stream>>>(cbuf, adaln_w + (long)l * DD * 6 * DD,
                                                   adaln_b + (long)l * 6 * DD, modb, 6 * DD);
        // attention branch
        k_rmsmod<<<TOK, 256, 0, stream>>>(hbuf, norm1_w + l * DD, modb, 6 * DD, 0, DD, hh);
        packT(qkv_w + (long)l * DD * 3 * DD, qkvT, DD, 3 * DD, 1);
        gemm(hh, DD, qkvT, 0, TOK, 3 * DD, DD, nullptr, nullptr, 0,
             qkvf, nullptr, 3 * DD, nullptr, 0, nullptr, 0, 0, nullptr, nullptr, 1, TOK / 128);
        k_qkvprep<<<TOK * NHH, 64, 0, stream>>>(qkvf, qn_w + l * HDD, kn_w + l * HDD,
                                                qb, kb, vtb);
        k_attn<<<BB * NHH, 256, 0, stream>>>(qb, kb, vtb, ob);
        packT(proj_w + (long)l * DD * DD, projT, DD, DD, 1);
        gemm(ob, DD, projT, 0, TOK, DD, DD, nullptr, nullptr, 0,
             nullptr, nullptr, DD, nullptr, 2, modb, 6 * DD, 2 * DD, hbuf, nullptr, 1, TOK / 128);
        // MoE branch
        k_rmsmod<<<TOK, 256, 0, stream>>>(hbuf, norm2_w + l * DD, modb, 6 * DD, 3 * DD, 4 * DD, hh);
        k_zero32<<<dim3(1), 256, 0, stream>>>(cnt, EE);
        k_zero32<<<dim3((EE * SLOTS) / 256), 256, 0, stream>>>(rowlist, EE * SLOTS);
        k_router<<<TOK / 8, 256, 0, stream>>>(hh, router_w + (long)l * DD * EE,
                                              cnt, rowlist, slotw);
        packT(moe_w1 + (long)l * EE * DD * HEE, w1T, DD, HEE, EE);
        packT(moe_w3 + (long)l * EE * DD * HEE, w3T, DD, HEE, EE);
        packT(moe_w2 + (long)l * EE * HEE * DD, w2T, HEE, DD, EE);
        k_moe_gemm1<<<dim3(HEE / 128, SLOTS / 128, EE), 256, 0, stream>>>(
            hh, w1T, w3T, cnt, rowlist, actb);
        gemm(actb, HEE, w2T, (long)DD * HEE, SLOTS, DD, HEE, cnt, rowlist, SLOTS,
             partial, nullptr, DD, nullptr, 3, nullptr, 0, 0, nullptr, slotw, EE, SLOTS / 128);
        k_moecombine<<<dim3((TOK * DD) / 256), 256, 0, stream>>>(hbuf, partial, modb);
    }

    // ---- final layer ----
    k_modmlp<<<dim3(6, BB), 256, 0, stream>>>(cbuf, fada_w, fada_b, fmodb, 2 * DD);
    k_rmsmod<<<TOK, 256, 0, stream>>>(hbuf, fnorm_w, fmodb, 2 * DD, 0, DD, hh);
    gemm(hh, DD, fpT, 0, TOK, DD, DD, nullptr, nullptr, 0,
         out, nullptr, DD, fproj_b, 0, nullptr, 0, 0, nullptr, nullptr, 1, TOK / 128);
}
